// TransformerBlock_87170656240146
// MI455X (gfx1250) — matrix operand order repaired and checked by reading, not compiled, not run on hardware
//
#include <hip/hip_runtime.h>
#include <hip/hip_bf16.h>
#include <math.h>

typedef __attribute__((ext_vector_type(16))) _Float16 v16h;
typedef __attribute__((ext_vector_type(8)))  _Float16 v8h;
typedef __attribute__((ext_vector_type(8)))  float    v8f;
typedef __attribute__((ext_vector_type(4)))  unsigned int u32x4;
typedef __attribute__((ext_vector_type(8)))  unsigned int u32x8;

#define D_MODEL 1024
#define NHEAD   16
#define DK      64
#define FFN_DIM 4096
#define SEQ     512
#define BATCH   16
#define ROWS    (BATCH * SEQ)   // 8192

// ---------------------------------------------------------------------------
// WMMA fragment helpers (CDNA5 v_wmma_f32_16x16x32_f16, wave32)
// ---------------------------------------------------------------------------

// A fragment: 16x32 (MxK). Lanes 0-15: row m, K = k0..k0+15;
// lanes 16-31: row m, K = k0+16..k0+31.
__device__ __forceinline__ v16h load_frag_a(const _Float16* __restrict__ A,
                                            int lane, long row, int k0, int ldk) {
  const int m   = lane & 15;
  const int sel = lane >> 4;
  const _Float16* p = A + (long)(row + m) * ldk + k0 + sel * 16;
  return *(const v16h*)p;           // K = k0 + sel*16 .. +15
}

// B fragment: 32x16 (KxN) from a row-major [N,K] ("Bt") array (global or LDS).
// Lanes 0-15: column n, K = k0..k0+15; lanes 16-31: column n, K = k0+16..k0+31.
__device__ __forceinline__ v16h load_frag_b(const _Float16* __restrict__ Bt,
                                            int lane, long col, int k0, int ldk) {
  const int n   = lane & 15;
  const int sel = lane >> 4;
  const _Float16* p = Bt + (long)(col + n) * ldk + k0 + sel * 16;
  return *(const v16h*)p;           // 32 contiguous bytes
}

__device__ __forceinline__ v8f wmma_f16(const v16h& a, const v16h& b, const v8f& c) {
  return __builtin_amdgcn_wmma_f32_16x16x32_f16(false, a, false, b, (short)0, c,
                                                false, false);
}

// ---------------------------------------------------------------------------
// Weight conversion: w[K,N] fp32 row-major -> out[N,K] f16 row-major (transpose)
// ---------------------------------------------------------------------------
__global__ void convert_w_f16t(const float* __restrict__ w, _Float16* __restrict__ out,
                               int K, int N) {
  long i = (long)blockIdx.x * blockDim.x + threadIdx.x;
  long total = (long)K * N;
  if (i >= total) return;
  long n = i / K, k = i - n * K;
  out[i] = (_Float16)w[k * N + n];
}

// ---------------------------------------------------------------------------
// LayerNorm: one 256-thread block per row of 1024 fp32; f16 output
// ---------------------------------------------------------------------------
__global__ __launch_bounds__(256)
void layernorm_f16(const float* __restrict__ x, const float* __restrict__ g,
                   const float* __restrict__ b, _Float16* __restrict__ out) {
  const int lane = threadIdx.x & 31;
  const int wid  = threadIdx.x >> 5;
  const long row = blockIdx.x;
  const float* xr = x + row * D_MODEL;

  float4 v = ((const float4*)xr)[threadIdx.x];       // 256 * 4 = 1024
  float s  = v.x + v.y + v.z + v.w;
  float sq = v.x * v.x + v.y * v.y + v.z * v.z + v.w * v.w;
#pragma unroll
  for (int m = 16; m >= 1; m >>= 1) {
    s  += __shfl_xor(s, m, 32);
    sq += __shfl_xor(sq, m, 32);
  }
  __shared__ float ssum[8], ssq[8];
  if (lane == 0) { ssum[wid] = s; ssq[wid] = sq; }
  __syncthreads();
  float tot = 0.f, totq = 0.f;
#pragma unroll
  for (int i = 0; i < 8; ++i) { tot += ssum[i]; totq += ssq[i]; }
  const float mu   = tot * (1.0f / D_MODEL);
  const float var  = totq * (1.0f / D_MODEL) - mu * mu;
  const float rstd = rsqrtf(var + 1e-5f);

  const int c = threadIdx.x * 4;
  float4 gg = ((const float4*)g)[threadIdx.x];
  float4 bb = ((const float4*)b)[threadIdx.x];
  _Float16* o = out + row * D_MODEL + c;
  o[0] = (_Float16)((v.x - mu) * rstd * gg.x + bb.x);
  o[1] = (_Float16)((v.y - mu) * rstd * gg.y + bb.y);
  o[2] = (_Float16)((v.z - mu) * rstd * gg.z + bb.z);
  o[3] = (_Float16)((v.w - mu) * rstd * gg.w + bb.w);
}

// ---------------------------------------------------------------------------
// Generic WMMA GEMM with TDM-staged B tiles.
//   C = A[M,K] @ Bt[N,K]^T, block tile 128x64 (8 waves, 32x32 per wave).
// B tile (64 N-rows x 64 K) is DMA'd global->LDS by the Tensor Data Mover
// (tensor_load_to_lds, TENSORcnt), double-buffered and overlapped with WMMAs.
// MODE 1: outF = acc + bias[col] + resid[row,col]   (fp32 out)
// MODE 2: outH = gelu_exact(acc + bias[col])        (f16 out)
// ---------------------------------------------------------------------------
template <int MODE>
__global__ __launch_bounds__(256)
void gemm_wmma(const _Float16* __restrict__ A, const _Float16* __restrict__ Bt,
               const float* __restrict__ bias, const float* __restrict__ resid,
               float* __restrict__ outF, _Float16* __restrict__ outH,
               int M, int N, int K) {
  const int lane  = threadIdx.x & 31;
  const int wid   = threadIdx.x >> 5;
  const int waveM = wid >> 1;          // 0..3
  const int waveN = wid & 1;           // 0..1
  const long row0 = (long)blockIdx.y * 128 + waveM * 32;
  const long col0 = (long)blockIdx.x * 64  + waveN * 32;

  __shared__ alignas(128) _Float16 btile[2][64][64];   // 2 x 8 KB

  // ---- Tensor DMA descriptor (ISA ch.8): 2D tile, 64 rows x 64 elems, 2B ----
  const unsigned long long gbase =
      (unsigned long long)(uintptr_t)Bt +
      (unsigned long long)blockIdx.x * 64ull * (unsigned long long)K * 2ull;
  const unsigned int ldsb = (unsigned int)(uintptr_t)&btile[0][0][0]; // LDS byte offset

  u32x8 g1;
  g1[0] = 0x00010000u;                        // wg_mask=0 (no cluster), data_size=2B
  g1[1] = ((unsigned int)K & 0xffffu) << 16;  // tensor_dim0 lo
  g1[2] = ((unsigned int)K >> 16) | (64u << 16); // tensor_dim0 hi | tensor_dim1 lo
  g1[3] = 64u << 16;                          // tile_dim0 = 64
  g1[4] = 64u;                                // tile_dim1 = 64, tile_dim2 = 0
  g1[5] = (unsigned int)K;                    // tensor_dim0_stride lo
  g1[6] = 0u;                                 // stride hi / dim1_stride lo
  g1[7] = 0u;

  const int nchunks = K >> 6;                 // K / 64

#define TDM_ISSUE(CHUNK)                                                        \
  do {                                                                          \
    if (wid == 0) {                                                             \
      unsigned long long ga = gbase + (unsigned long long)(CHUNK) * 128ull;     \
      u32x4 g0;                                                                 \
      g0[0] = 1u;                                /* count=1, user D# */         \
      g0[1] = ldsb + (unsigned int)((CHUNK) & 1) * (64u * 64u * 2u);            \
      g0[2] = (unsigned int)ga;                                                 \
      g0[3] = (unsigned int)((ga >> 32) & 0x1ffffffu) | (2u << 30); /* type=2 */\
      asm volatile("tensor_load_to_lds %0, %1" ::"s"(g0), "s"(g1) : "memory");  \
    }                                                                           \
  } while (0)

  TDM_ISSUE(0);
  if (wid == 0) __builtin_amdgcn_s_wait_tensorcnt(0);
  __syncthreads();

  v8f c[2][2] = {};
  for (int ch = 0; ch < nchunks; ++ch) {
    if (ch + 1 < nchunks) {
      TDM_ISSUE(ch + 1);                      // overlaps with this chunk's math
      __builtin_prefetch(A + (row0 + (lane & 15)) * (long)K + (ch + 1) * 64, 0, 0);
    }
    const _Float16* bt = &btile[ch & 1][0][0];
#pragma unroll
    for (int kk = 0; kk < 64; kk += 32) {
      const int k0 = ch * 64 + kk;
      v16h a0 = load_frag_a(A, lane, row0,      k0, K);
      v16h a1 = load_frag_a(A, lane, row0 + 16, k0, K);
      v16h b0 = load_frag_b(bt, lane, waveN * 32,      kk, 64);  // ds_load
      v16h b1 = load_frag_b(bt, lane, waveN * 32 + 16, kk, 64);
      c[0][0] = wmma_f16(a0, b0, c[0][0]);
      c[0][1] = wmma_f16(a0, b1, c[0][1]);
      c[1][0] = wmma_f16(a1, b0, c[1][0]);
      c[1][1] = wmma_f16(a1, b1, c[1][1]);
    }
    if (ch + 1 < nchunks) {
      if (wid == 0) __builtin_amdgcn_s_wait_tensorcnt(0);
      __syncthreads();                        // publish next tile / retire reads
    }
  }
#undef TDM_ISSUE

  const int n  = lane & 15;
  const int mb = (lane >> 4) * 8;
#pragma unroll
  for (int mt = 0; mt < 2; ++mt)
#pragma unroll
    for (int nt = 0; nt < 2; ++nt) {
      const long col = col0 + nt * 16 + n;
#pragma unroll
      for (int r = 0; r < 8; ++r) {
        const long row = row0 + mt * 16 + mb + r;
        float v = c[mt][nt][r];
        if constexpr (MODE == 1) {
          outF[row * N + col] = v + bias[col] + resid[row * N + col];
        } else {
          float t = v + bias[col];
          outH[row * N + col] =
              (_Float16)(0.5f * t * (1.0f + erff(t * 0.70710678118654752f)));
        }
      }
    }
}

// ---------------------------------------------------------------------------
// QKV GEMM: z = 0/1/2 selects W_{q,k,v}; permuted stores:
//   q -> [B,H,T,DK] (scaled 1/sqrt(DK)), k -> [B,H,T,DK], v -> [B,H,DK,T] (transposed)
// ---------------------------------------------------------------------------
__global__ __launch_bounds__(256)
void gemm_qkv(const _Float16* __restrict__ A,
              const _Float16* __restrict__ wq16, const _Float16* __restrict__ wk16,
              const _Float16* __restrict__ wv16,
              _Float16* __restrict__ q, _Float16* __restrict__ k,
              _Float16* __restrict__ vT) {
  const int z = blockIdx.z;
  const _Float16* Bt = (z == 0) ? wq16 : (z == 1) ? wk16 : wv16;
  const int lane  = threadIdx.x & 31;
  const int wid   = threadIdx.x >> 5;
  const long row0 = (long)blockIdx.y * 128 + (wid >> 1) * 32;
  const long col0 = (long)blockIdx.x * 64  + (wid & 1) * 32;

  v8f c[2][2] = {};
  for (int k0 = 0; k0 < D_MODEL; k0 += 32) {
    v16h a0 = load_frag_a(A, lane, row0,      k0, D_MODEL);
    v16h a1 = load_frag_a(A, lane, row0 + 16, k0, D_MODEL);
    v16h b0 = load_frag_b(Bt, lane, col0,      k0, D_MODEL);
    v16h b1 = load_frag_b(Bt, lane, col0 + 16, k0, D_MODEL);
    c[0][0] = wmma_f16(a0, b0, c[0][0]);
    c[0][1] = wmma_f16(a0, b1, c[0][1]);
    c[1][0] = wmma_f16(a1, b0, c[1][0]);
    c[1][1] = wmma_f16(a1, b1, c[1][1]);
  }

  const int n  = lane & 15;
  const int mb = (lane >> 4) * 8;
#pragma unroll
  for (int mt = 0; mt < 2; ++mt)
#pragma unroll
    for (int nt = 0; nt < 2; ++nt)
#pragma unroll
      for (int r = 0; r < 8; ++r) {
        const long row = row0 + mt * 16 + mb + r;   // token index
        const long col = col0 + nt * 16 + n;        // feature index
        const long b  = row >> 9, t = row & (SEQ - 1);
        const long h  = col >> 6, d = col & (DK - 1);
        const long bh = b * NHEAD + h;
        const float v = c[mt][nt][r];
        if (z == 0)      q[(bh * SEQ + t) * DK + d] = (_Float16)(v * 0.125f);
        else if (z == 1) k[(bh * SEQ + t) * DK + d] = (_Float16)v;
        else             vT[(bh * DK + d) * SEQ + t] = (_Float16)v;
      }
}

// ---------------------------------------------------------------------------
// Causal flash attention. Block = 4 waves; wave w handles 16 query rows
// (qb = bx*64 + w*16) of head blockIdx.y. Key chunks of 32, online softmax.
// Output ctx[b, t, h*64+d] f16.
// ---------------------------------------------------------------------------
__global__ __launch_bounds__(128)
void attention_kernel(const _Float16* __restrict__ q, const _Float16* __restrict__ k,
                      const _Float16* __restrict__ vT, _Float16* __restrict__ ctx) {
  const int lane = threadIdx.x & 31;
  const int wid  = threadIdx.x >> 5;               // 0..3
  const long bh  = blockIdx.y;                     // 0..255
  const int  qb  = blockIdx.x * 64 + wid * 16;     // query tile base

  const _Float16* Q  = q  + bh * SEQ * DK;
  const _Float16* Kk = k  + bh * SEQ * DK;
  const _Float16* Vt = vT + bh * DK * SEQ;

  __shared__ alignas(32) _Float16 pbuf[4][16][32]; // per-wave P tile

  // Q row tile as two A fragments (d = 0..31 and 32..63); Q pre-scaled.
  const v16h aq0 = load_frag_a(Q, lane, qb, 0,  DK);
  const v16h aq1 = load_frag_a(Q, lane, qb, 32, DK);

  v8f o[4] = {};                                   // d tiles of 16
  float mrun[8], lrun[8];
#pragma unroll
  for (int r = 0; r < 8; ++r) { mrun[r] = -1e30f; lrun[r] = 0.0f; }

  const int n  = lane & 15;
  const int mb = (lane >> 4) * 8;

  for (int kb = 0; kb < qb + 16; kb += 32) {
    // S = Q @ K^T for 32 keys: two 16-col C fragments, 2 K-steps each
    v8f s0 = {}, s1 = {};
    s0 = wmma_f16(aq0, load_frag_b(Kk, lane, kb,      0,  DK), s0);
    s0 = wmma_f16(aq1, load_frag_b(Kk, lane, kb,      32, DK), s0);
    s1 = wmma_f16(aq0, load_frag_b(Kk, lane, kb + 16, 0,  DK), s1);
    s1 = wmma_f16(aq1, load_frag_b(Kk, lane, kb + 16, 32, DK), s1);

    const int col0 = kb + n, col1 = col0 + 16;
#pragma unroll
    for (int r = 0; r < 8; ++r) {
      const int qrow = qb + mb + r;
      if (col0 > qrow) s0[r] = -1e30f;             // causal mask
      if (col1 > qrow) s1[r] = -1e30f;

      float mx = fmaxf(s0[r], s1[r]);
#pragma unroll
      for (int m = 1; m <= 8; m <<= 1) mx = fmaxf(mx, __shfl_xor(mx, m, 32));
      const float mnew  = fmaxf(mrun[r], mx);
      const float alpha = __expf(mrun[r] - mnew);
      mrun[r] = mnew;
      const float p0 = __expf(s0[r] - mnew);
      const float p1 = __expf(s1[r] - mnew);
      float rs = p0 + p1;
#pragma unroll
      for (int m = 1; m <= 8; m <<= 1) rs += __shfl_xor(rs, m, 32);
      lrun[r] = lrun[r] * alpha + rs;
#pragma unroll
      for (int dt = 0; dt < 4; ++dt) o[dt][r] *= alpha;
      pbuf[wid][mb + r][n]      = (_Float16)p0;    // wave-private LDS stage
      pbuf[wid][mb + r][n + 16] = (_Float16)p1;
    }
    // Same-wave DS ops are in-order: safe to read the staged P tile back.
    const v16h ap = load_frag_a(&pbuf[wid][0][0], lane, 0, 0, 32);
#pragma unroll
    for (int dt = 0; dt < 4; ++dt)
      o[dt] = wmma_f16(ap, load_frag_b(Vt, lane, dt * 16, kb, SEQ), o[dt]);
  }

  // normalize and store ctx[b*512+t, h*64+d]
  const long rowb = (bh >> 4) * SEQ + qb + mb;
  const long colb = (bh & 15) * DK;
#pragma unroll
  for (int r = 0; r < 8; ++r) {
    const float inv = 1.0f / lrun[r];
#pragma unroll
    for (int dt = 0; dt < 4; ++dt)
      ctx[(rowb + r) * D_MODEL + colb + dt * 16 + n] = (_Float16)(o[dt][r] * inv);
  }
}

// ---------------------------------------------------------------------------
// Host-side orchestration
// ---------------------------------------------------------------------------
extern "C" void kernel_launch(void* const* d_in, const int* in_sizes, int n_in,
                              void* d_out, int out_size, void* d_ws, size_t ws_size,
                              hipStream_t stream) {
  (void)in_sizes; (void)n_in; (void)out_size; (void)ws_size;
  const float* x    = (const float*)d_in[0];
  const float* ln1g = (const float*)d_in[1];
  const float* ln1b = (const float*)d_in[2];
  const float* wq   = (const float*)d_in[3];
  const float* wk   = (const float*)d_in[4];
  const float* wv   = (const float*)d_in[5];
  const float* wo   = (const float*)d_in[6];
  const float* bo   = (const float*)d_in[7];
  const float* ln2g = (const float*)d_in[8];
  const float* ln2b = (const float*)d_in[9];
  const float* w1   = (const float*)d_in[10];
  const float* b1   = (const float*)d_in[11];
  const float* w2   = (const float*)d_in[12];
  const float* b2   = (const float*)d_in[13];
  float* out = (float*)d_out;

  char* ws = (char*)d_ws;
  size_t off = 0;
  auto take = [&](size_t bytes) { void* p = ws + off; off += bytes; return p; };
  const size_t W   = (size_t)D_MODEL * D_MODEL * 2;     // 2 MB
  const size_t ACT = (size_t)ROWS * D_MODEL * 2;        // 16 MB

  _Float16* wq16 = (_Float16*)take(W);
  _Float16* wk16 = (_Float16*)take(W);
  _Float16* wv16 = (_Float16*)take(W);
  _Float16* wo16 = (_Float16*)take(W);
  _Float16* w1t  = (_Float16*)take((size_t)D_MODEL * FFN_DIM * 2);
  _Float16* w2t  = (_Float16*)take((size_t)FFN_DIM * D_MODEL * 2);
  _Float16* h16  = (_Float16*)take(ACT);                // LN1 out, reused as ctx
  _Float16* q16  = (_Float16*)take(ACT);                // reused as LN2 out
  _Float16* k16  = (_Float16*)take(ACT);
  _Float16* vT16 = (_Float16*)take(ACT);
  float*    x1   = (float*)take((size_t)ROWS * D_MODEL * 4);
  _Float16* a16  = (_Float16*)take((size_t)ROWS * FFN_DIM * 2);

  // 1) weight conversion (fp32 [K,N] -> f16 [N,K])
  const int CT = 256;
  convert_w_f16t<<<(D_MODEL * D_MODEL + CT - 1) / CT, CT, 0, stream>>>(wq, wq16, D_MODEL, D_MODEL);
  convert_w_f16t<<<(D_MODEL * D_MODEL + CT - 1) / CT, CT, 0, stream>>>(wk, wk16, D_MODEL, D_MODEL);
  convert_w_f16t<<<(D_MODEL * D_MODEL + CT - 1) / CT, CT, 0, stream>>>(wv, wv16, D_MODEL, D_MODEL);
  convert_w_f16t<<<(D_MODEL * D_MODEL + CT - 1) / CT, CT, 0, stream>>>(wo, wo16, D_MODEL, D_MODEL);
  convert_w_f16t<<<(D_MODEL * FFN_DIM + CT - 1) / CT, CT, 0, stream>>>(w1, w1t, D_MODEL, FFN_DIM);
  convert_w_f16t<<<(FFN_DIM * D_MODEL + CT - 1) / CT, CT, 0, stream>>>(w2, w2t, FFN_DIM, D_MODEL);

  // 2) LN1
  layernorm_f16<<<ROWS, 256, 0, stream>>>(x, ln1g, ln1b, h16);

  // 3) QKV (z selects weight + store permutation)
  gemm_qkv<<<dim3(D_MODEL / 64, ROWS / 128, 3), 256, 0, stream>>>(
      h16, wq16, wk16, wv16, q16, k16, vT16);

  // 4) causal attention -> ctx (reuses h16)
  attention_kernel<<<dim3(SEQ / 64, BATCH * NHEAD), 128, 0, stream>>>(q16, k16, vT16, h16);

  // 5) O-projection + bias + residual(x) -> x1 (fp32)
  gemm_wmma<1><<<dim3(D_MODEL / 64, ROWS / 128), 256, 0, stream>>>(
      h16, wo16, bo, x, x1, nullptr, ROWS, D_MODEL, D_MODEL);

  // 6) LN2 -> h2 (reuses q16)
  layernorm_f16<<<ROWS, 256, 0, stream>>>(x1, ln2g, ln2b, q16);

  // 7) FFN1 + bias + exact GELU -> a16
  gemm_wmma<2><<<dim3(FFN_DIM / 64, ROWS / 128), 256, 0, stream>>>(
      q16, w1t, b1, nullptr, nullptr, a16, ROWS, FFN_DIM, D_MODEL);

  // 8) FFN2 + bias + residual(x1) -> out (fp32)
  gemm_wmma<1><<<dim3(D_MODEL / 64, ROWS / 128), 256, 0, stream>>>(
      a16, w2t, b2, x1, out, nullptr, ROWS, D_MODEL, FFN_DIM);
}
